// Encoder_biGRU_4964982194749
// MI455X (gfx1250) — matrix operand order repaired and checked by reading, not compiled, not run on hardware
//
#include <hip/hip_runtime.h>
#include <hip/hip_fp16.h>

#define SEQ   4096
#define EMB   200
#define EMBP  224        // 7 * 32, padded K for f16 WMMA
#define HID   256
#define G3    768        // 3*HID
#define KSTEPS 7

typedef __attribute__((ext_vector_type(16))) _Float16 v16h;
typedef __attribute__((ext_vector_type(8)))  float    v8f;

union V16HU { v16h v; uint4 q[2]; };

// Async global->LDS prefetch path (gfx1250); guarded so either toolchain compiles.
#if defined(__gfx1250__) &&                                        \
    __has_builtin(__builtin_amdgcn_global_load_async_to_lds_b32) && \
    __has_builtin(__builtin_amdgcn_s_wait_asynccnt)
#define USE_ASYNC_LDS 1
typedef __attribute__((address_space(1))) int glb_i32;   // builtin wants int* (b32)
typedef __attribute__((address_space(3))) int lds_i32;
#else
#define USE_ASYNC_LDS 0
#endif

// ---- workspace layout (bytes) ----
#define XH_OFF   0ull
#define W1H_OFF  (XH_OFF  + (size_t)SEQ * EMBP * 2)
#define W2H_OFF  (W1H_OFF + (size_t)G3  * EMBP * 2)
#define GI1_OFF  (W2H_OFF + (size_t)G3  * EMBP * 2)
#define GI2_OFF  (GI1_OFF + (size_t)SEQ * G3   * 4)

// ---- output layout (float elements) ----
#define OUT_HID  ((size_t)SEQ * EMB)                 // after embeddings
#define OUT_H1   (OUT_HID + (size_t)SEQ * 2 * HID)   // after hiddens
#define OUT_H2   (OUT_H1 + HID)

// ------------------------------------------------------------------
// 1) Embedding gather: writes f32 embeddings output + f16 padded X
// ------------------------------------------------------------------
__global__ void k_gather(const int* __restrict__ utt,
                         const float* __restrict__ embw,
                         float* __restrict__ out_emb,
                         _Float16* __restrict__ xh) {
    int idx = blockIdx.x * blockDim.x + threadIdx.x;
    if (idx >= SEQ * EMBP) return;
    int t = idx / EMBP, e = idx % EMBP;
    float v = 0.f;
    if (e < EMB) {
        v = embw[(size_t)utt[t] * EMB + e];
        out_emb[(size_t)t * EMB + e] = v;
    }
    xh[(size_t)t * EMBP + e] = (_Float16)v;
}

// ------------------------------------------------------------------
// 2) Convert/pad W_ih (both directions) to f16 [768 x 224]
// ------------------------------------------------------------------
__global__ void k_convw(const float* __restrict__ w1,
                        const float* __restrict__ w2,
                        _Float16* __restrict__ w1h,
                        _Float16* __restrict__ w2h) {
    int idx = blockIdx.x * blockDim.x + threadIdx.x;
    if (idx >= G3 * EMBP) return;
    int row = idx / EMBP, e = idx % EMBP;
    const float* w = blockIdx.y ? w2 : w1;
    _Float16* wh   = blockIdx.y ? w2h : w1h;
    wh[(size_t)row * EMBP + e] =
        (e < EMB) ? (_Float16)w[(size_t)row * EMB + e] : (_Float16)0.f;
}

// ------------------------------------------------------------------
// 3) WMMA GEMM: GI[dir] = X(4096x224,f16) @ W_ih[dir]^T + b_ih[dir]
//    Each wave computes a 16(M) x 64(N) strip: A reused across 4 B tiles.
// ------------------------------------------------------------------
__global__ void __launch_bounds__(128)
k_wmma_gemm(const _Float16* __restrict__ xh,
            const _Float16* __restrict__ w1h,
            const _Float16* __restrict__ w2h,
            const float* __restrict__ b1,
            const float* __restrict__ b2,
            float* __restrict__ gi1,
            float* __restrict__ gi2) {
    const int dir = blockIdx.y;
    const _Float16* wf  = dir ? w2h : w1h;
    const float*   bias = dir ? b2  : b1;
    float*         gi   = dir ? gi2 : gi1;

    const int wave = (blockIdx.x * blockDim.x + threadIdx.x) >> 5;
    const int lane = threadIdx.x & 31;
    const int mTile  = wave / 12;   // 0..255
    const int nStrip = wave % 12;   // 0..11 (64 columns each)
    const int mBase = mTile * 16;
    const int nBase = nStrip * 64;

    const int lmod = lane & 15;
    const int lhi  = lane >> 4;     // 0 or 1

    const _Float16* aRow = xh + (size_t)(mBase + lmod) * EMBP;

    v8f acc[4];
    #pragma unroll
    for (int j = 0; j < 4; ++j) { v8f z = {}; acc[j] = z; }

    for (int ks = 0; ks < KSTEPS; ++ks) {
        const int kBase = ks * 32;
        V16HU a;
        const int alow = kBase + lhi * 8;
        a.q[0] = *(const uint4*)(aRow + alow);
        a.q[1] = *(const uint4*)(aRow + alow + 16);
        #pragma unroll
        for (int j = 0; j < 4; ++j) {
            const _Float16* bp =
                wf + (size_t)(nBase + j * 16 + lmod) * EMBP + kBase + lhi * 8;
            V16HU b;
            b.q[0] = ((const uint4*)bp)[0];
            b.q[1] = ((const uint4*)bp)[2];
            acc[j] = __builtin_amdgcn_wmma_f32_16x16x32_f16(
                         false, a.v, false, b.v, (short)0, acc[j], false, false);
        }
    }

    #pragma unroll
    for (int j = 0; j < 4; ++j) {
        const int n = nBase + j * 16 + lmod;
        const float bv = bias[n];
        #pragma unroll
        for (int r = 0; r < 8; ++r) {
            const int m = mBase + r + 8 * lhi;   // C layout: vgpr r -> M = r + 8*(lane>=16)
            gi[(size_t)m * G3 + n] = acc[j][r] + bv;
        }
    }
}

// ------------------------------------------------------------------
// 4) Persistent biGRU recurrence: one block per direction (2 WGPs).
//    768 threads; thread = one row of W_hh, held in 128 packed-f16 VGPRs.
//    Per step: 128 v_pk_fma_f16 dot + LDS gate exchange + gating (tid<256).
//    GI[t] for the NEXT step is double-buffered into LDS with
//    GLOBAL_LOAD_ASYNC_TO_LDS_B32 so the L2 round-trip overlaps the matvec.
// ------------------------------------------------------------------
__global__ void __launch_bounds__(768, 1)
k_gru(const float* __restrict__ whh1, const float* __restrict__ whh2,
      const float* __restrict__ bhh1, const float* __restrict__ bhh2,
      const float* __restrict__ gi1,  const float* __restrict__ gi2,
      const float* __restrict__ h01,  const float* __restrict__ h02,
      float* __restrict__ out) {
    const int dir = blockIdx.x;
    const float* whh = dir ? whh2 : whh1;
    const float* bhh = dir ? bhh2 : bhh1;
    const float* gi  = dir ? gi2  : gi1;
    const float* h0  = dir ? h02  : h01;
    float* out_hid  = out + OUT_HID;
    float* out_last = out + (dir ? OUT_H2 : OUT_H1);

    __shared__ __half2 hsh[HID / 2];   // packed h for the dot
    __shared__ float   hf[HID];        // f32 h for the z*h term
    __shared__ float   gh[G3];         // gate pre-activations (hidden side)
#if USE_ASYNC_LDS
    __shared__ float   gibuf[2][G3];   // double-buffered GI[t] prefetch
#endif

    const int tid = threadIdx.x;

    // Preload this thread's W_hh row as 128 packed f16 pairs (VGPR-resident).
    __half2 wreg[HID / 2];
    {
        const float* wr = whh + (size_t)tid * HID;
        #pragma unroll
        for (int i = 0; i < HID / 2; ++i)
            wreg[i] = __floats2half2_rn(wr[2 * i], wr[2 * i + 1]);
    }
    const float bh = bhh[tid];

    if (tid < HID) {
        const float v = h0[tid];
        hf[tid] = v;
        ((__half*)hsh)[tid] = __float2half(v);
    }

#if USE_ASYNC_LDS
    // Kick off the prefetch for step 0 (one float per thread -> full GI row).
    {
        const int t0 = dir ? (SEQ - 1) : 0;
        __builtin_amdgcn_global_load_async_to_lds_b32(
            (glb_i32*)(gi + (size_t)t0 * G3 + tid),
            (lds_i32*)&gibuf[0][tid], 0, 0);
    }
#endif
    __syncthreads();

    for (int s = 0; s < SEQ; ++s) {
        const int t = dir ? (SEQ - 1 - s) : s;

#if USE_ASYNC_LDS
        // Prefetch next step's GI row into the other buffer; it overlaps the
        // matvec below.  Slot (s+1)&1 was last read in step s-1 and is free
        // (end-of-step barrier already passed).
        if (s + 1 < SEQ) {
            const int tn = dir ? (SEQ - 2 - s) : (s + 1);
            __builtin_amdgcn_global_load_async_to_lds_b32(
                (glb_i32*)(gi + (size_t)tn * G3 + tid),
                (lds_i32*)&gibuf[(s + 1) & 1][tid], 0, 0);
        }
#endif

        // gh[row] = dot(w_hh[row], h) + b_hh[row]
        __half2 acc2 = __floats2half2_rn(0.f, 0.f);
        #pragma unroll
        for (int i = 0; i < HID / 2; ++i)
            acc2 = __hfma2(wreg[i], hsh[i], acc2);
        gh[tid] = __low2float(acc2) + __high2float(acc2) + bh;

#if USE_ASYNC_LDS
        // Async completions are in-order: allow the newest prefetch to remain
        // outstanding, but require this step's buffer to have landed.
        if (s + 1 < SEQ) __builtin_amdgcn_s_wait_asynccnt(1);
        else             __builtin_amdgcn_s_wait_asynccnt(0);
#endif
        __syncthreads();

        if (tid < HID) {
#if USE_ASYNC_LDS
            const float* git = gibuf[s & 1];
#else
            const float* git = gi + (size_t)t * G3;
#endif
            const float ir  = git[tid];
            const float iz  = git[HID + tid];
            const float in_ = git[2 * HID + tid];
            const float r  = 1.f / (1.f + __expf(-(ir + gh[tid])));
            const float z  = 1.f / (1.f + __expf(-(iz + gh[HID + tid])));
            const float n  = tanhf(in_ + r * gh[2 * HID + tid]);
            const float hn = (1.f - z) * n + z * hf[tid];
            hf[tid] = hn;
            ((__half*)hsh)[tid] = __float2half(hn);
            out_hid[(size_t)t * (2 * HID) + dir * HID + tid] = hn;
        }
        __syncthreads();
    }

    if (tid < HID) out_last[tid] = hf[tid];
}

// ------------------------------------------------------------------
extern "C" void kernel_launch(void* const* d_in, const int* in_sizes, int n_in,
                              void* d_out, int out_size, void* d_ws, size_t ws_size,
                              hipStream_t stream) {
    const int*   utt  = (const int*)d_in[0];
    const float* h01  = (const float*)d_in[1];
    const float* h02  = (const float*)d_in[2];
    const float* embw = (const float*)d_in[3];
    const float* wih1 = (const float*)d_in[4];
    const float* whh1 = (const float*)d_in[5];
    const float* bih1 = (const float*)d_in[6];
    const float* bhh1 = (const float*)d_in[7];
    const float* wih2 = (const float*)d_in[8];
    const float* whh2 = (const float*)d_in[9];
    const float* bih2 = (const float*)d_in[10];
    const float* bhh2 = (const float*)d_in[11];

    float* out = (float*)d_out;
    char*  ws  = (char*)d_ws;
    _Float16* xh  = (_Float16*)(ws + XH_OFF);
    _Float16* w1h = (_Float16*)(ws + W1H_OFF);
    _Float16* w2h = (_Float16*)(ws + W2H_OFF);
    float* gi1 = (float*)(ws + GI1_OFF);
    float* gi2 = (float*)(ws + GI2_OFF);

    k_gather<<<(SEQ * EMBP + 255) / 256, 256, 0, stream>>>(utt, embw, out, xh);
    k_convw<<<dim3((G3 * EMBP + 255) / 256, 2), 256, 0, stream>>>(wih1, wih2, w1h, w2h);
    // 256 M-tiles x 12 N-strips = 3072 waves per direction; 4 waves/block
    k_wmma_gemm<<<dim3(768, 2), 128, 0, stream>>>(xh, w1h, w2h, bih1, bih2, gi1, gi2);
    k_gru<<<2, 768, 0, stream>>>(whh1, whh2, bhh1, bhh2, gi1, gi2, h01, h02, out);
}
